// MultiHeadAttention_47450798686410
// MI455X (gfx1250) — compile-verified, matrix operand order checked by reading, not run on hardware
//
#include <hip/hip_runtime.h>
#include <hip/hip_bf16.h>

typedef __attribute__((ext_vector_type(16))) _Float16 v16h;
typedef __attribute__((ext_vector_type(8)))  _Float16 v8h;
typedef __attribute__((ext_vector_type(4)))  _Float16 v4h;
typedef __attribute__((ext_vector_type(8)))  float    v8f;
typedef __attribute__((ext_vector_type(4)))  unsigned int v4u;
typedef __attribute__((ext_vector_type(4)))  int v4i;
typedef __attribute__((ext_vector_type(8)))  int v8i;

#define S_LEN 2048
#define NHEAD 6
#define HDIM  64
#define EMB   384

#define USE_ASYNC_LDS 1
#define USE_TDM 1

#if USE_TDM && __has_builtin(__builtin_amdgcn_tensor_load_to_lds) && \
    __has_builtin(__builtin_amdgcn_s_wait_tensorcnt)
#define HAVE_TDM 1
#else
#define HAVE_TDM 0
#endif

// ---- fragment load: source is K-major contiguous per lane-row.
// v16h elements 0..7  = k: kbase + 8*hf .. +7   (16 contiguous bytes)
// v16h elements 8..15 = k: kbase + 16 + 8*hf .. (16 contiguous bytes)
__device__ __forceinline__ v16h frag_load_k8(const _Float16* p, int ld, int row,
                                             int kbase, int hf) {
    const _Float16* base = p + row * ld + kbase + 8 * hf;
    v8h lo = *(const v8h*)(base);
    v8h hi = *(const v8h*)(base + 16);
    return __builtin_shufflevector(lo, hi, 0, 1, 2, 3, 4, 5, 6, 7,
                                   8, 9, 10, 11, 12, 13, 14, 15);
}

__device__ __forceinline__ float rowmax16(float x) {
#pragma unroll
    for (int m = 1; m < 16; m <<= 1) x = fmaxf(x, __shfl_xor(x, m, 32));
    return x;
}
__device__ __forceinline__ float rowsum16(float x) {
#pragma unroll
    for (int m = 1; m < 16; m <<= 1) x += __shfl_xor(x, m, 32);
    return x;
}

__device__ __forceinline__ unsigned lds_off32(const void* p) {
    return (unsigned)(uintptr_t)p;  // low 32 bits of flat LDS address = LDS offset
}

#if USE_ASYNC_LDS
// memory -> LDS without VGPR round trip; tracked by ASYNCcnt (ISA 15.18.3)
__device__ __forceinline__ void async_b128(unsigned lds_off, const void* gaddr) {
    asm volatile("global_load_async_to_lds_b128 %0, %1, off"
                 :: "v"(lds_off), "v"(gaddr) : "memory");
}
__device__ __forceinline__ void wait_async0() {
    asm volatile("s_wait_asynccnt 0" ::: "memory");
}
#endif

#if HAVE_TDM
// Tensor Data Mover: DMA a 2D tile (16-bit elements) global -> LDS.
// D# packing per ISA 8.3-8.6; tensor dims == tile dims so no OOB clipping.
__device__ __forceinline__ void tdm_load_2d(unsigned lds_addr, const void* gaddr,
                                            unsigned tile_d0, unsigned tile_d1,
                                            unsigned long long stride0) {
    unsigned long long ga = (unsigned long long)(uintptr_t)gaddr;
    v4u g0;
    g0[0] = 1u;                                              // count=1 (user D#)
    g0[1] = lds_addr;                                        // lds_addr (bytes)
    g0[2] = (unsigned)(ga & 0xffffffffu);                    // global_addr[31:0]
    g0[3] = (unsigned)((ga >> 32) & 0x1ffffffu) | (2u << 30);// addr[56:32] | type=2
    v8i g1;
    g1[0] = (int)(1u << 16);                                 // data_size=1 (2 bytes)
    g1[1] = (int)((tile_d0 & 0xffffu) << 16);                // tensor_dim0[15:0]@bit48
    g1[2] = (int)((tile_d0 >> 16) & 0xffffu) |               // tensor_dim0[31:16]
            (int)((tile_d1 & 0xffffu) << 16);                // tensor_dim1[15:0]
    g1[3] = (int)((tile_d1 >> 16) & 0xffffu) |               // tensor_dim1[31:16]
            (int)((tile_d0 & 0xffffu) << 16);                // tile_dim0
    g1[4] = (int)(tile_d1 & 0xffffu);                        // tile_dim1 (tile_dim2=0)
    g1[5] = (int)(stride0 & 0xffffffffu);                    // tensor_dim0_stride[31:0]
    g1[6] = (int)((stride0 >> 32) & 0xffffu);                // stride0[47:32], stride1=0
    g1[7] = 0;
    v4i gz4 = {0, 0, 0, 0};                                  // groups 2/3: dims 2..4 unused
    v8i gz8 = {0, 0, 0, 0, 0, 0, 0, 0};
    __builtin_amdgcn_tensor_load_to_lds(g0, g1, gz4, gz4, gz8, 0);
}
#endif

// ---------------- Kernel 1: QKV projection
// Q,K f16 [B,H,S,D]; V f16 transposed [B,H,D,S] (so P@V B-operand is row-contig)
__global__ __launch_bounds__(128)
void qkv_proj_kernel(const float* __restrict__ x,
                     const float* __restrict__ Wq,
                     const float* __restrict__ Wk,
                     const float* __restrict__ Wv,
                     _Float16* __restrict__ Qh,
                     _Float16* __restrict__ Kh,
                     _Float16* __restrict__ Vh) {
    __shared__ alignas(16) _Float16 xs[64 * 32];   // [row][k]
    __shared__ alignas(16) _Float16 wt[64 * 32];   // transposed: [n][k]
    const int tid  = threadIdx.x;
    const int lane = tid & 31, wv = tid >> 5;
    const int hf = (lane >> 4) & 1, l16 = lane & 15;
    const int rbase = blockIdx.x * 64;
    const int cblk  = blockIdx.y;            // 0..17
    const int wsel  = cblk / 6;              // 0:Q 1:K 2:V
    const int cm    = (cblk % 6) * 64;       // column base inside matrix
    const float* W  = (wsel == 0) ? Wq : (wsel == 1) ? Wk : Wv;

    v8f acc[4] = {};
    for (int kk = 0; kk < EMB; kk += 32) {
        // stage x tile 64x32 (f32 -> f16); static 4 iterations per thread
#pragma unroll
        for (int it = 0; it < 4; ++it) {
            int i = tid * 4 + it * 512;
            int r = i >> 5, c = i & 31;
            float4 xv = *(const float4*)&x[(size_t)(rbase + r) * EMB + kk + c];
            v4h h4 = {(_Float16)xv.x, (_Float16)xv.y, (_Float16)xv.z, (_Float16)xv.w};
            *(v4h*)&xs[i] = h4;
        }
        // stage W tile 32x64 transposed -> wt[n][k]; static 4 iterations
#pragma unroll
        for (int it = 0; it < 4; ++it) {
            int i = tid * 4 + it * 512;
            int k = i >> 6, n = i & 63;
            float4 w4 = *(const float4*)&W[(size_t)(kk + k) * EMB + cm + n];
            wt[(n + 0) * 32 + k] = (_Float16)w4.x;
            wt[(n + 1) * 32 + k] = (_Float16)w4.y;
            wt[(n + 2) * 32 + k] = (_Float16)w4.z;
            wt[(n + 3) * 32 + k] = (_Float16)w4.w;
        }
        __syncthreads();
        v16h a = frag_load_k8(xs, 32, wv * 16 + l16, 0, hf);
#pragma unroll
        for (int t = 0; t < 4; ++t) {
            v16h bf = frag_load_k8(wt, 32, t * 16 + l16, 0, hf);
            acc[t] = __builtin_amdgcn_wmma_f32_16x16x32_f16(
                false, a, false, bf, (short)0, acc[t], false, false);
        }
        __syncthreads();
    }

    if (wsel == 2) {
        // V: [B,H,D,S]; the 8 rows of a C-register group are consecutive s -> packed store
        int rg0 = rbase + wv * 16 + 8 * hf;
        int b = rg0 / S_LEN, s0 = rg0 % S_LEN;
#pragma unroll
        for (int t = 0; t < 4; ++t) {
            int cg = cm + t * 16 + l16;
            int h = cg / HDIM, d = cg % HDIM;
            v8h pk = {(_Float16)acc[t][0], (_Float16)acc[t][1], (_Float16)acc[t][2],
                      (_Float16)acc[t][3], (_Float16)acc[t][4], (_Float16)acc[t][5],
                      (_Float16)acc[t][6], (_Float16)acc[t][7]};
            *(v8h*)&Vh[(((size_t)b * NHEAD + h) * HDIM + d) * S_LEN + s0] = pk;
        }
    } else {
        _Float16* Out = (wsel == 0) ? Qh : Kh;
#pragma unroll
        for (int t = 0; t < 4; ++t) {
            int cg = cm + t * 16 + l16;
            int h = cg / HDIM, d = cg % HDIM;
#pragma unroll
            for (int r = 0; r < 8; ++r) {
                int rg = rbase + wv * 16 + r + 8 * hf;
                int b = rg / S_LEN, s = rg % S_LEN;
                Out[(((size_t)b * NHEAD + h) * S_LEN + s) * HDIM + d] = (_Float16)acc[t][r];
            }
        }
    }
}

// ---------------- Kernel 2: causal flash attention; O f16 [B,S,E]
__global__ __launch_bounds__(128)
void flash_attn_kernel(const _Float16* __restrict__ Qh,
                       const _Float16* __restrict__ Kh,
                       const _Float16* __restrict__ Vh,   // [B,H,D,S]
                       _Float16* __restrict__ Oh) {
    __shared__ alignas(16) _Float16 kt[32 * 64];    // [key][d]
    __shared__ alignas(16) _Float16 vtT[64 * 32];   // [d][key]
    __shared__ alignas(16) _Float16 pls[4][16 * 32];
    const int tid  = threadIdx.x;
    const int lane = tid & 31, wv = tid >> 5;
    const int hf = (lane >> 4) & 1, l16 = lane & 15;
    const int bh    = blockIdx.y;
    const int bb    = bh / NHEAD, hh = bh % NHEAD;
    const int qblk  = blockIdx.x * 64;
    const int qbase = qblk + wv * 16;
    const size_t hbase = (size_t)bh * S_LEN * HDIM;

    // Q fragments for d=0..31 and d=32..63 (4 x b128 global loads)
    const _Float16* qp = Qh + hbase + (size_t)(qbase + l16) * HDIM + 8 * hf;
    v16h qa0, qa1;
    {
        v8h a = *(const v8h*)(qp + 0);
        v8h b = *(const v8h*)(qp + 16);
        v8h c = *(const v8h*)(qp + 32);
        v8h d = *(const v8h*)(qp + 48);
        qa0 = __builtin_shufflevector(a, b, 0,1,2,3,4,5,6,7,8,9,10,11,12,13,14,15);
        qa1 = __builtin_shufflevector(c, d, 0,1,2,3,4,5,6,7,8,9,10,11,12,13,14,15);
    }

    v8f o[4] = {};
    float mrow[8], lrow[8];
#pragma unroll
    for (int r = 0; r < 8; ++r) { mrow[r] = -1e30f; lrow[r] = 0.f; }

    const float scale = 0.125f;  // 1/sqrt(64)
    const int jend = qblk + 64;

    for (int j = 0; j < jend; j += 32) {
        // ---- stage K tile 32x64 (contiguous) and V^T tile 64x32 (rows strided by S)
#if HAVE_TDM
        if (wv == 0) {
            // one TDM descriptor per tile; TENSORcnt tracks completion
            tdm_load_2d(lds_off32(kt), Kh + hbase + (size_t)j * HDIM,
                        /*d0=*/64, /*d1=*/32, /*stride0=*/64);
            tdm_load_2d(lds_off32(vtT), Vh + (size_t)bh * HDIM * S_LEN + j,
                        /*d0=*/32, /*d1=*/64, /*stride0=*/S_LEN);
            __builtin_amdgcn_s_wait_tensorcnt(0);
        }
#elif USE_ASYNC_LDS
        {
            const _Float16* kg = Kh + hbase + (size_t)j * HDIM;
#pragma unroll
            for (int it = 0; it < 2; ++it) {
                int i = tid + it * 128;
                async_b128(lds_off32(kt + i * 8), kg + i * 8);
            }
#pragma unroll
            for (int it = 0; it < 2; ++it) {
                int i = tid + it * 128;
                int row = i >> 2, ch = i & 3;
                const _Float16* vg = Vh + ((size_t)bh * HDIM + row) * S_LEN + j + ch * 8;
                async_b128(lds_off32(vtT + row * 32 + ch * 8), vg);
            }
            wait_async0();
        }
#else
        {
            const uint4* ks = (const uint4*)(Kh + hbase + (size_t)j * HDIM);
            uint4* kd = (uint4*)kt;
#pragma unroll
            for (int it = 0; it < 2; ++it) {
                int i = tid + it * 128;
                kd[i] = ks[i];
            }
#pragma unroll
            for (int it = 0; it < 2; ++it) {
                int i = tid + it * 128;
                int row = i >> 2, ch = i & 3;
                *(uint4*)(vtT + row * 32 + ch * 8) =
                    *(const uint4*)(Vh + ((size_t)bh * HDIM + row) * S_LEN + j + ch * 8);
            }
        }
#endif
        __syncthreads();

        // scores: two 16-key n-tiles; K-dim 64 = two wmma each
        v8f c0 = {}, c1 = {};
        {
            v16h b00 = frag_load_k8(kt, 64, l16, 0, hf);
            v16h b01 = frag_load_k8(kt, 64, l16, 32, hf);
            c0 = __builtin_amdgcn_wmma_f32_16x16x32_f16(false, qa0, false, b00, (short)0, c0, false, false);
            c0 = __builtin_amdgcn_wmma_f32_16x16x32_f16(false, qa1, false, b01, (short)0, c0, false, false);
            v16h b10 = frag_load_k8(kt, 64, 16 + l16, 0, hf);
            v16h b11 = frag_load_k8(kt, 64, 16 + l16, 32, hf);
            c1 = __builtin_amdgcn_wmma_f32_16x16x32_f16(false, qa0, false, b10, (short)0, c1, false, false);
            c1 = __builtin_amdgcn_wmma_f32_16x16x32_f16(false, qa1, false, b11, (short)0, c1, false, false);
        }

        // causal mask + online softmax
        float bm[8];
#pragma unroll
        for (int r = 0; r < 8; ++r) {
            int mg  = qbase + r + 8 * hf;
            int kg0 = j + l16, kg1 = j + 16 + l16;
            float s0 = c0[r] * scale; if (kg0 > mg) s0 = -1e30f;
            float s1 = c1[r] * scale; if (kg1 > mg) s1 = -1e30f;
            c0[r] = s0; c1[r] = s1;
            bm[r] = rowmax16(fmaxf(s0, s1));
        }
        _Float16* pw = pls[wv];
#pragma unroll
        for (int r = 0; r < 8; ++r) {
            float nm   = fmaxf(mrow[r], bm[r]);
            float corr = __expf(mrow[r] - nm);
            mrow[r] = nm;
            float p0 = __expf(c0[r] - nm);
            float p1 = __expf(c1[r] - nm);
            lrow[r] = lrow[r] * corr + rowsum16(p0 + p1);
#pragma unroll
            for (int t = 0; t < 4; ++t) o[t][r] *= corr;
            pw[(r + 8 * hf) * 32 + l16]      = (_Float16)p0;
            pw[(r + 8 * hf) * 32 + 16 + l16] = (_Float16)p1;
        }
        // same-wave LDS store->load is in-order; relayout P into A fragment
        v16h pa = frag_load_k8(pw, 32, l16, 0, hf);
#pragma unroll
        for (int t = 0; t < 4; ++t) {
            v16h vb = frag_load_k8(vtT, 32, t * 16 + l16, 0, hf);
            o[t] = __builtin_amdgcn_wmma_f32_16x16x32_f16(
                false, pa, false, vb, (short)0, o[t], false, false);
        }
        __syncthreads();
    }

    // normalize + store O f16 row-major [B,S,E]
#pragma unroll
    for (int r = 0; r < 8; ++r) {
        float inv = 1.0f / lrow[r];
        int rg = qbase + r + 8 * hf;
#pragma unroll
        for (int t = 0; t < 4; ++t) {
            int d = t * 16 + l16;
            Oh[((size_t)bb * S_LEN + rg) * EMB + hh * HDIM + d] = (_Float16)(o[t][r] * inv);
        }
    }
}

// ---------------- Kernel 3: output projection O f16 @ Wo + bo -> fp32
__global__ __launch_bounds__(128)
void out_proj_kernel(const _Float16* __restrict__ Oh,
                     const float* __restrict__ Wo,
                     const float* __restrict__ bo,
                     float* __restrict__ out) {
    __shared__ alignas(16) _Float16 os[64 * 32];   // [row][k]
    __shared__ alignas(16) _Float16 wt[64 * 32];   // transposed: [n][k]
    const int tid  = threadIdx.x;
    const int lane = tid & 31, wv = tid >> 5;
    const int hf = (lane >> 4) & 1, l16 = lane & 15;
    const int rbase = blockIdx.x * 64;
    const int cbase = blockIdx.y * 64;

    v8f acc[4] = {};
    for (int kk = 0; kk < EMB; kk += 32) {
        // stage O tile 64x32; static 2 iterations (v8h copies)
#pragma unroll
        for (int it = 0; it < 2; ++it) {
            int i = tid * 8 + it * 1024;
            int r = i >> 5, c = i & 31;
            *(v8h*)&os[i] = *(const v8h*)&Oh[(size_t)(rbase + r) * EMB + kk + c];
        }
        // stage Wo tile 32x64 transposed; static 4 iterations
#pragma unroll
        for (int it = 0; it < 4; ++it) {
            int i = tid * 4 + it * 512;
            int k = i >> 6, n = i & 63;
            float4 w4 = *(const float4*)&Wo[(size_t)(kk + k) * EMB + cbase + n];
            wt[(n + 0) * 32 + k] = (_Float16)w4.x;
            wt[(n + 1) * 32 + k] = (_Float16)w4.y;
            wt[(n + 2) * 32 + k] = (_Float16)w4.z;
            wt[(n + 3) * 32 + k] = (_Float16)w4.w;
        }
        __syncthreads();
        v16h a = frag_load_k8(os, 32, wv * 16 + l16, 0, hf);
#pragma unroll
        for (int t = 0; t < 4; ++t) {
            v16h bf = frag_load_k8(wt, 32, t * 16 + l16, 0, hf);
            acc[t] = __builtin_amdgcn_wmma_f32_16x16x32_f16(
                false, a, false, bf, (short)0, acc[t], false, false);
        }
        __syncthreads();
    }
#pragma unroll
    for (int t = 0; t < 4; ++t) {
        int c = cbase + t * 16 + l16;
        float bias = bo[c];
#pragma unroll
        for (int r = 0; r < 8; ++r) {
            int rg = rbase + wv * 16 + r + 8 * hf;
            out[(size_t)rg * EMB + c] = acc[t][r] + bias;
        }
    }
}

extern "C" void kernel_launch(void* const* d_in, const int* in_sizes, int n_in,
                              void* d_out, int out_size, void* d_ws, size_t ws_size,
                              hipStream_t stream) {
    const float* x  = (const float*)d_in[0];
    const float* Wq = (const float*)d_in[1];
    const float* Wk = (const float*)d_in[2];
    const float* Wv = (const float*)d_in[3];
    const float* Wo = (const float*)d_in[4];
    const float* bo = (const float*)d_in[5];
    float* out = (float*)d_out;

    const size_t N = (size_t)8 * S_LEN * EMB;  // 6,291,456 elements
    _Float16* Qh = (_Float16*)d_ws;
    _Float16* Kh = Qh + N;
    _Float16* Vh = Kh + N;   // transposed [B,H,D,S]
    _Float16* Oh = Vh + N;

    dim3 blk(128);
    qkv_proj_kernel<<<dim3(256, 18), blk, 0, stream>>>(x, Wq, Wk, Wv, Qh, Kh, Vh);
    flash_attn_kernel<<<dim3(S_LEN / 64, 8 * NHEAD), blk, 0, stream>>>(Qh, Kh, Vh, Oh);
    out_proj_kernel<<<dim3(256, 6), blk, 0, stream>>>(Oh, Wo, bo, out);
}
